// MSAPairWeightedAveraging_2061584302843
// MI455X (gfx1250) — matrix operand order repaired and checked by reading, not compiled, not run on hardware
//
#include <hip/hip_runtime.h>
#include <hip/hip_bf16.h>
#include <float.h>

// ---------------------------------------------------------------------------
// MSA pair-weighted averaging for MI455X (gfx1250, wave32, WMMA bf16).
// Workspace layout (bytes, total ~196 MB):
//   [0)           Vt   bf16 [h][s][d][j]        8*256*32*512  = 33,554,432 el
//   [64 MiB)      Gate bf16 [s][i][h*32+d]      256*512*256   = 33,554,432 el
//   [128 MiB)     Watt bf16 [h][i][j]           8*512*512     =  2,097,152 el
//   [132 MiB)     Inner bf16 [s][i][h*32+d]     256*512*256   = 33,554,432 el
// ---------------------------------------------------------------------------

typedef __bf16 bf16;
typedef __attribute__((ext_vector_type(16))) __bf16 v16bf;
typedef __attribute__((ext_vector_type(8)))  float  v8f;

#define DIM_MSA   64
#define DIM_PAIR  128
#define DIM_HEAD  32
#define HEADS     8
#define DIM_INNER 256
#define S_LEN     256
#define N_LEN     512

// ---------------------------------------------------------------------------
// K1: LN(msa) @ w_vg  -> values (transposed bf16) + sigmoid gates (bf16)
// One block per (s,j) row; 256 threads; thread t computes value col t and
// gate col t (+256 in w_vg).
// ---------------------------------------------------------------------------
__global__ void __launch_bounds__(256)
k1_msa_ln_vg(const float* __restrict__ msa,
             const float* __restrict__ ln_g, const float* __restrict__ ln_b,
             const float* __restrict__ w_vg,
             bf16* __restrict__ vt, bf16* __restrict__ gate)
{
    const int row = blockIdx.x;          // row = s*512 + j
    const int s = row >> 9;
    const int j = row & (N_LEN - 1);
    const int t = threadIdx.x;

    __shared__ float xs[DIM_MSA];
    __shared__ float red[DIM_MSA];
    __shared__ float xhat[DIM_MSA];
    __shared__ float stats[2];

    if (t < DIM_MSA) xs[t] = msa[(size_t)row * DIM_MSA + t];
    __syncthreads();

    if (t < DIM_MSA) red[t] = xs[t];
    __syncthreads();
    for (int off = 32; off >= 1; off >>= 1) {
        if (t < off) red[t] += red[t + off];
        __syncthreads();
    }
    if (t == 0) stats[0] = red[0] * (1.0f / DIM_MSA);
    __syncthreads();
    const float mu = stats[0];

    if (t < DIM_MSA) { float d = xs[t] - mu; red[t] = d * d; }
    __syncthreads();
    for (int off = 32; off >= 1; off >>= 1) {
        if (t < off) red[t] += red[t + off];
        __syncthreads();
    }
    if (t == 0) stats[1] = rsqrtf(red[0] * (1.0f / DIM_MSA) + 1e-5f);
    __syncthreads();
    const float rstd = stats[1];

    if (t < DIM_MSA) xhat[t] = (xs[t] - mu) * rstd * ln_g[t] + ln_b[t];
    __syncthreads();

    float accv = 0.0f, accg = 0.0f;
    const int cv = t;            // value column 0..255
    const int cg = t + 256;      // gate column 256..511
    #pragma unroll 8
    for (int k = 0; k < DIM_MSA; ++k) {
        const float xh = xhat[k];
        accv = fmaf(xh, w_vg[k * 512 + cv], accv);
        accg = fmaf(xh, w_vg[k * 512 + cg], accg);
    }
    const int h = cv >> 5, d = cv & 31;
    // Vt[h][s][d][j]
    vt[(((size_t)h * S_LEN + s) * DIM_HEAD + d) * N_LEN + j] = (bf16)accv;
    // Gate[s][j(=i)][h*32+d]
    const float gv = 1.0f / (1.0f + __expf(-accg));
    gate[(size_t)row * DIM_INNER + cv] = (bf16)gv;
}

// ---------------------------------------------------------------------------
// K2: LN(pair) @ w_b -> masked softmax over j -> Watt bf16 [h][i][j]
// One block per row i; 256 threads; thread handles j=t and j=t+256.
// ---------------------------------------------------------------------------
__global__ void __launch_bounds__(256)
k2_pair_softmax(const float* __restrict__ pair,
                const float* __restrict__ ln_g, const float* __restrict__ ln_b,
                const float* __restrict__ w_b,
                const unsigned char* __restrict__ mask,
                bf16* __restrict__ watt)
{
    const int i = blockIdx.x;
    const int t = threadIdx.x;
    __shared__ float logit[HEADS][N_LEN];   // 16 KB

    #pragma unroll
    for (int rep = 0; rep < 2; ++rep) {
        const int j = t + rep * 256;
        const float* x = pair + ((size_t)i * N_LEN + j) * DIM_PAIR;
        float sum = 0.0f, sumsq = 0.0f;
        for (int c = 0; c < DIM_PAIR; ++c) { const float v = x[c]; sum += v; sumsq += v * v; }
        const float mu   = sum * (1.0f / DIM_PAIR);
        const float var  = sumsq * (1.0f / DIM_PAIR) - mu * mu;
        const float rstd = rsqrtf(var + 1e-5f);
        float acc[HEADS];
        #pragma unroll
        for (int h = 0; h < HEADS; ++h) acc[h] = 0.0f;
        for (int c = 0; c < DIM_PAIR; ++c) {
            const float xh = (x[c] - mu) * rstd * ln_g[c] + ln_b[c];
            #pragma unroll
            for (int h = 0; h < HEADS; ++h) acc[h] = fmaf(xh, w_b[c * HEADS + h], acc[h]);
        }
        const bool m = mask[j] != 0;
        #pragma unroll
        for (int h = 0; h < HEADS; ++h) logit[h][j] = m ? acc[h] : -FLT_MAX;
    }
    __syncthreads();

    // one wave per head (8 waves of 32 lanes)
    const int h    = t >> 5;
    const int lane = t & 31;
    float mx = -FLT_MAX;
    for (int j = lane; j < N_LEN; j += 32) mx = fmaxf(mx, logit[h][j]);
    #pragma unroll
    for (int off = 16; off >= 1; off >>= 1) mx = fmaxf(mx, __shfl_xor(mx, off, 32));
    float ssum = 0.0f;
    for (int j = lane; j < N_LEN; j += 32) {
        const float e = __expf(logit[h][j] - mx);
        logit[h][j] = e;
        ssum += e;
    }
    #pragma unroll
    for (int off = 16; off >= 1; off >>= 1) ssum += __shfl_xor(ssum, off, 32);
    const float inv = 1.0f / ssum;
    for (int j = lane; j < N_LEN; j += 32)
        watt[((size_t)h * N_LEN + i) * N_LEN + j] = (bf16)(logit[h][j] * inv);
}

// ---------------------------------------------------------------------------
// K3: out[h,s,i,d] = sum_j W[h,i,j] * V[h,s,j,d] via WMMA bf16, fused gating.
// grid = (8 i-groups of 64 rows, 256 s); 8 waves per block, wave = head.
// Each wave processes FOUR 16-row i-tiles so each B fragment pair (V data)
// is reused by 4 A fragments: 6 KB of L2 reads per 8 WMMAs (~21.8 FLOP/B),
// halving L2 traffic vs one-tile-per-block.
// Fragments use contiguous K order (k = 16*half + s) for both A and B.
// ---------------------------------------------------------------------------
#define ITILES 4

__global__ void __launch_bounds__(256)
k3_weighted_avg(const bf16* __restrict__ watt,
                const bf16* __restrict__ vt,
                const bf16* __restrict__ gate,
                bf16* __restrict__ inner)
{
    const int ig    = blockIdx.x;          // 0..7  (group of 64 i rows)
    const int s     = blockIdx.y;          // 0..255
    const int t     = threadIdx.x;
    const int h     = t >> 5;              // wave id == head
    const int lane  = t & 31;
    const int half  = lane >> 4;           // 0: lanes 0-15, 1: lanes 16-31
    const int l16   = lane & 15;
    const int i0    = ig * (16 * ITILES);

    v8f acc0[ITILES];
    v8f acc1[ITILES];
    #pragma unroll
    for (int it = 0; it < ITILES; ++it) { acc0[it] = (v8f){}; acc1[it] = (v8f){}; }

    // A row base per i-tile for this lane (row mi = i0 + it*16 + l16)
    const bf16* arow[ITILES];
    #pragma unroll
    for (int it = 0; it < ITILES; ++it)
        arow[it] = watt + ((size_t)h * N_LEN + (i0 + it * 16 + l16)) * N_LEN;

    const bf16* vbase = vt + ((size_t)h * S_LEN + s) * (DIM_HEAD * N_LEN);

    for (int kj = 0; kj < 16; ++kj) {
        const int j0 = kj * 32;

        // B fragments: 32x16 bf16.  lane column n=l16; K half per lane group.
        // Vt is d-major so each lane reads 16 contiguous bf16 (32 bytes).
        v16bf b0, b1;
        {
            const bf16* p = vbase + (size_t)l16 * N_LEN + j0 + half * 16;
            #pragma unroll
            for (int q = 0; q < 16; ++q) b0[q] = p[q];
        }
        {
            const bf16* p = vbase + (size_t)(16 + l16) * N_LEN + j0 + half * 16;
            #pragma unroll
            for (int q = 0; q < 16; ++q) b1[q] = p[q];
        }

        #pragma unroll
        for (int it = 0; it < ITILES; ++it) {
            // A fragment: 16x32 bf16. lanes 0-15: K 0..15;
            //                         lanes 16-31: K 16..31
            v16bf a;
            const bf16* p0 = arow[it] + j0 + half * 16;
            const bf16* p1 = p0 + 8;
            #pragma unroll
            for (int q = 0; q < 8; ++q) { a[q] = p0[q]; a[q + 8] = p1[q]; }

            acc0[it] = __builtin_amdgcn_wmma_f32_16x16x32_bf16(
                false, a, false, b0, (short)0, acc0[it], false, false);
            acc1[it] = __builtin_amdgcn_wmma_f32_16x16x32_bf16(
                false, a, false, b1, (short)0, acc1[it], false, false);
        }
    }

    // C/D layout: VGPR p -> M = p + half*8, N = l16. Apply gate, store bf16.
    #pragma unroll
    for (int it = 0; it < ITILES; ++it) {
        #pragma unroll
        for (int p = 0; p < 8; ++p) {
            const int m = p + half * 8;
            const int i = i0 + it * 16 + m;
            const size_t base = ((size_t)s * N_LEN + i) * DIM_INNER + h * DIM_HEAD;
            const float g0 = (float)gate[base + l16];
            const float g1 = (float)gate[base + 16 + l16];
            inner[base + l16]      = (bf16)(acc0[it][p] * g0);
            inner[base + 16 + l16] = (bf16)(acc1[it][p] * g1);
        }
    }
}

// ---------------------------------------------------------------------------
// K4: out = inner @ w_out   (131072 x 256 x 64), WMMA bf16, fp32 out.
// Block = 4 sequential 16-row tiles, 4 waves; wave w owns output columns
// [16w, 16w+16).  w_out staged ONCE per block into LDS as bf16 [k][n] and
// reused by all four row-tiles (amortizes the 64 KB L2 staging read 4x).
// ---------------------------------------------------------------------------
#define RTILES 4

__global__ void __launch_bounds__(128)
k4_out_proj(const bf16* __restrict__ inner,
            const float* __restrict__ w_out,
            float* __restrict__ out)
{
    const int btile = blockIdx.x;          // 0..2047
    const int t     = threadIdx.x;         // 128 threads
    const int wave  = t >> 5;              // 0..3 -> column chunk
    const int lane  = t & 31;
    const int half  = lane >> 4;
    const int l16   = lane & 15;

    __shared__ bf16 wb[DIM_INNER * DIM_MSA];   // 256x64 bf16 = 32 KB
    for (int idx = t; idx < DIM_INNER * DIM_MSA; idx += 128)
        wb[idx] = (bf16)w_out[idx];
    __syncthreads();

    for (int rt = 0; rt < RTILES; ++rt) {
        const int r0 = (btile * RTILES + rt) * 16;
        const bf16* arow = inner + (size_t)(r0 + l16) * DIM_INNER;

        v8f acc = {};
        #pragma unroll
        for (int kk = 0; kk < 8; ++kk) {
            const int k0 = kk * 32;
            v16bf a;
            {
                const bf16* p0 = arow + k0 + half * 16;
                const bf16* p1 = p0 + 8;
                #pragma unroll
                for (int q = 0; q < 8; ++q) { a[q] = p0[q]; a[q + 8] = p1[q]; }
            }
            v16bf b;
            {
                const bf16* p = wb + (k0 + half * 16) * DIM_MSA + wave * 16 + l16;
                #pragma unroll
                for (int q = 0; q < 16; ++q) b[q] = p[q * DIM_MSA];
            }
            acc = __builtin_amdgcn_wmma_f32_16x16x32_bf16(false, a, false, b,
                                                          (short)0, acc, false, false);
        }

        #pragma unroll
        for (int p = 0; p < 8; ++p) {
            const int r = r0 + p + half * 8;
            out[(size_t)r * DIM_MSA + wave * 16 + l16] = acc[p];
        }
    }
}

// ---------------------------------------------------------------------------
extern "C" void kernel_launch(void* const* d_in, const int* in_sizes, int n_in,
                              void* d_out, int out_size, void* d_ws, size_t ws_size,
                              hipStream_t stream)
{
    const float*         msa      = (const float*)d_in[0];
    const float*         pair     = (const float*)d_in[1];
    const unsigned char* mask     = (const unsigned char*)d_in[2];
    const float*         ln_msa_g = (const float*)d_in[3];
    const float*         ln_msa_b = (const float*)d_in[4];
    const float*         w_vg     = (const float*)d_in[5];
    const float*         ln_p_g   = (const float*)d_in[6];
    const float*         ln_p_b   = (const float*)d_in[7];
    const float*         w_b      = (const float*)d_in[8];
    const float*         w_out    = (const float*)d_in[9];
    float* out = (float*)d_out;

    char* ws = (char*)d_ws;
    bf16* vt    = (bf16*)(ws);                                       // 64 MiB
    bf16* gate  = (bf16*)(ws + (size_t)64 * 1024 * 1024);            // 64 MiB
    bf16* watt  = (bf16*)(ws + (size_t)128 * 1024 * 1024);           //  4 MiB
    bf16* inner = (bf16*)(ws + (size_t)132 * 1024 * 1024);           // 64 MiB

    k1_msa_ln_vg<<<S_LEN * N_LEN, 256, 0, stream>>>(msa, ln_msa_g, ln_msa_b,
                                                    w_vg, vt, gate);
    k2_pair_softmax<<<N_LEN, 256, 0, stream>>>(pair, ln_p_g, ln_p_b, w_b,
                                               mask, watt);
    dim3 g3(N_LEN / (16 * ITILES), S_LEN);
    k3_weighted_avg<<<g3, 256, 0, stream>>>(watt, vt, gate, inner);
    k4_out_proj<<<(S_LEN * N_LEN) / (16 * RTILES), 128, 0, stream>>>(inner, w_out, out);
}
